// WirelessMemory_46050639347844
// MI455X (gfx1250) — matrix operand order repaired and checked by reading, not compiled, not run on hardware
//
#include <hip/hip_runtime.h>

// ---------------------------------------------------------------------------
// Memory-bank NCE step, fused single-pass over the 200k x 256 bank.
//  traffic floor: 204.8 MB read + 204.8 MB write  @ 23.3 TB/s  ~ 17.6 us
//  GEMM 6.55 GFLOP done as bf16-split (hi+lo) WMMA -> far below compute roof.
// ---------------------------------------------------------------------------

#define DDIM      256
#define BQ        64
#define NROWS     200000
#define TILE      16
#define NT_TILES  (NROWS / TILE)   // 12500
#define GRID_MAIN 1250             // 10 tiles per block
#define TEMP_INV  20.0f            // 1 / 0.05
#define LSE_SHIFT 20.0f            // |logit| <= 20 since rows are unit norm
#define EPSV      1e-12f

typedef __attribute__((ext_vector_type(16))) __bf16 v16bf;
typedef __attribute__((ext_vector_type(8)))  __bf16 v8bf;
typedef __attribute__((ext_vector_type(4)))  __bf16 v4bf;
typedef __attribute__((ext_vector_type(8)))  float  v8f;
typedef __attribute__((ext_vector_type(4)))  float  v4f;

__device__ __forceinline__ float block_reduce_256(float v, float* red) {
  const int tid = threadIdx.x;
  red[tid] = v;
  __syncthreads();
#pragma unroll
  for (int s = 128; s > 0; s >>= 1) {
    if (tid < s) red[tid] += red[tid + s];
    __syncthreads();
  }
  const float r = red[0];
  __syncthreads();
  return r;
}

// ---- kernel 1: normalize queries, emit f32 + bf16 hi/lo split -------------
__global__ __launch_bounds__(256) void nce_normalize(
    const float* __restrict__ inputs, float* __restrict__ ws_x,
    __bf16* __restrict__ xhi, __bf16* __restrict__ xlo) {
  __shared__ float red[256];
  const int b = blockIdx.x, tid = threadIdx.x;
  const float v  = inputs[b * DDIM + tid];
  const float ss = block_reduce_256(v * v, red);
  const float xn = v / fmaxf(sqrtf(ss), EPSV);
  ws_x[b * DDIM + tid] = xn;
  const __bf16 h = (__bf16)xn;
  xhi[b * DDIM + tid] = h;
  xlo[b * DDIM + tid] = (__bf16)(xn - (float)h);
}

// ---- kernel 2: stream bank: copy-out + bf16-split WMMA logits + exp-sums ---
__global__ __launch_bounds__(128) void nce_main(
    const float* __restrict__ feat, const __bf16* __restrict__ xhi,
    const __bf16* __restrict__ xlo, float* __restrict__ out_feat /* d_out+1 */,
    float* __restrict__ partial) {
  __shared__ __align__(32) __bf16 fh[TILE][DDIM];  // tile hi split
  __shared__ __align__(32) __bf16 fl[TILE][DDIM];  // tile lo split

  const int tid  = threadIdx.x;
  const int lane = tid & 31;
  const int wv   = tid >> 5;            // wave 0..3 -> M-tile (16 batch rows)
  const int hi16 = (lane >= 16) ? 1 : 0;

  // A fragments (constant across tiles): contiguous 16-bit A 16x32 order used here:
  // lanes 0-15 hold K 0..15, lanes 16-31 hold K 16..31 (matches B order).
  const int mrow = wv * 16 + (lane & 15);
  const int kofA = hi16 ? 16 : 0;
  v16bf ah[8], al[8];
#pragma unroll
  for (int kb = 0; kb < 8; ++kb) {
    const int k0 = kb * 32 + kofA;
    v8bf h0 = *(const v8bf*)(xhi + mrow * DDIM + k0);
    v8bf h1 = *(const v8bf*)(xhi + mrow * DDIM + k0 + 8);
    v8bf l0 = *(const v8bf*)(xlo + mrow * DDIM + k0);
    v8bf l1 = *(const v8bf*)(xlo + mrow * DDIM + k0 + 8);
    ah[kb] = __builtin_shufflevector(h0, h1, 0,1,2,3,4,5,6,7,8,9,10,11,12,13,14,15);
    al[kb] = __builtin_shufflevector(l0, l1, 0,1,2,3,4,5,6,7,8,9,10,11,12,13,14,15);
  }

  // B fragment addressing: 16-bit B 32x16: lanes 0-15 = K 0..15 (col n=lane),
  // lanes 16-31 = K 16..31 -> 16 contiguous bf16 per lane per K-block.
  const int nrow = lane & 15;
  const int kofB = hi16 ? 16 : 0;

  float acc[8];
#pragma unroll
  for (int r = 0; r < 8; ++r) acc[r] = 0.0f;

  for (int t = blockIdx.x; t < NT_TILES; t += GRID_MAIN) {
    const int n0 = t * TILE;
    __syncthreads();  // previous tile's WMMA reads of LDS are done
    // stage: copy tile to output (bulk of the copy) + split to LDS
#pragma unroll
    for (int i = 0; i < 8; ++i) {
      const int idx = i * 128 + tid;          // 1024 float4 units per tile
      const int row = idx >> 6;
      const int c   = (idx & 63) * 4;
      const v4f v = __builtin_nontemporal_load(
          (const v4f*)(feat + (size_t)(n0 + row) * DDIM + c));
      float* o = out_feat + (size_t)(n0 + row) * DDIM + c;  // 4B-aligned only
      __builtin_nontemporal_store(v.x, o + 0);
      __builtin_nontemporal_store(v.y, o + 1);
      __builtin_nontemporal_store(v.z, o + 2);
      __builtin_nontemporal_store(v.w, o + 3);
      const __bf16 hx = (__bf16)v.x, hy = (__bf16)v.y;
      const __bf16 hz = (__bf16)v.z, hw = (__bf16)v.w;
      v4bf hv = {hx, hy, hz, hw};
      v4bf lv = {(__bf16)(v.x - (float)hx), (__bf16)(v.y - (float)hy),
                 (__bf16)(v.z - (float)hz), (__bf16)(v.w - (float)hw)};
      *(v4bf*)&fh[row][c] = hv;
      *(v4bf*)&fl[row][c] = lv;
    }
    __syncthreads();

    // logits tile: C[16x16] = x[Mtile] . feat_tile^T  (split-bf16, 3 WMMAs/K)
    v8f c8 = {};
#pragma unroll
    for (int kb = 0; kb < 8; ++kb) {
      const int k0 = kb * 32 + kofB;
      v16bf bh = *(const v16bf*)&fh[nrow][k0];
      v16bf bl = *(const v16bf*)&fl[nrow][k0];
      c8 = __builtin_amdgcn_wmma_f32_16x16x32_bf16(false, al[kb], false, bh,
                                                   (short)0, c8, false, false);
      c8 = __builtin_amdgcn_wmma_f32_16x16x32_bf16(false, ah[kb], false, bl,
                                                   (short)0, c8, false, false);
      c8 = __builtin_amdgcn_wmma_f32_16x16x32_bf16(false, ah[kb], false, bh,
                                                   (short)0, c8, false, false);
    }
    // shifted exp; accumulate per-lane (defer lane reduction to block end)
#pragma unroll
    for (int r = 0; r < 8; ++r)
      acc[r] += __expf(c8[r] * TEMP_INV - LSE_SHIFT);
  }

  // reduce across the 16 lanes holding each batch row (C layout: n = lane&15,
  // m = r + 8*(lane>=16)); xor masks 1..8 stay inside each 16-lane group.
#pragma unroll
  for (int r = 0; r < 8; ++r) {
#pragma unroll
    for (int m = 8; m > 0; m >>= 1) acc[r] += __shfl_xor(acc[r], m, 32);
  }
  if ((lane & 15) == 0) {
    const int mbase = wv * 16 + (hi16 ? 8 : 0);
#pragma unroll
    for (int r = 0; r < 8; ++r)
      partial[(size_t)blockIdx.x * BQ + mbase + r] = acc[r];
  }
}

// ---- kernel 3: reduce partials, loss, momentum-update target rows ----------
__global__ __launch_bounds__(256) void nce_finalize(
    const float* __restrict__ feat, const float* __restrict__ scores,
    const int* __restrict__ targets, const float* __restrict__ ws_x,
    const float* __restrict__ partial, float* __restrict__ d_out) {
  __shared__ float red[256];
  __shared__ float Ssum[BQ];
  const int tid = threadIdx.x;

  {  // S[b] = sum_g partial[g*64+b]   (fixed-order, deterministic)
    const int b = tid & 63, part = tid >> 6;
    float s = 0.0f;
    for (int g = part; g < GRID_MAIN; g += 4) s += partial[(size_t)g * BQ + b];
    red[tid] = s;
    __syncthreads();
    if (tid < BQ)
      Ssum[tid] = red[tid] + red[tid + 64] + red[tid + 128] + red[tid + 192];
    __syncthreads();
  }

  float loss = 0.0f;
  for (int b = 0; b < BQ; ++b) {  // sequential -> duplicate targets: last wins
    const int   t  = targets[b];
    const float sc = scores[b];
    const float xv = ws_x[b * DDIM + tid];
    const float fv = feat[(size_t)t * DDIM + tid];
    const float dot = block_reduce_256(xv * fv, red);  // exact f32 target dot
    const float u   = 0.2f * fv + 0.8f * sc * xv;      // momentum update
    const float nsq = block_reduce_256(u * u, red);
    d_out[1 + (size_t)t * DDIM + tid] = u / fmaxf(sqrtf(nsq), EPSV);
    const float lse = LSE_SHIFT + __logf(Ssum[b]);
    loss += sc * (lse - dot * TEMP_INV);
  }
  if (tid == 0) d_out[0] = loss;
}

extern "C" void kernel_launch(void* const* d_in, const int* in_sizes, int n_in,
                              void* d_out, int out_size, void* d_ws,
                              size_t ws_size, hipStream_t stream) {
  (void)in_sizes; (void)n_in; (void)out_size; (void)ws_size;
  const float* inputs   = (const float*)d_in[0];
  const float* scores   = (const float*)d_in[1];
  const float* features = (const float*)d_in[2];
  const int*   targets  = (const int*)d_in[3];
  float* out = (float*)d_out;

  char*   ws      = (char*)d_ws;
  float*  ws_x    = (float*)(ws);             // 64*256*4  = 65536 B
  __bf16* xhi     = (__bf16*)(ws + 65536);    // 64*256*2  = 32768 B
  __bf16* xlo     = (__bf16*)(ws + 98304);    // 64*256*2  = 32768 B
  float*  partial = (float*)(ws + 131072);    // 1250*64*4 = 320000 B

  nce_normalize<<<BQ, 256, 0, stream>>>(inputs, ws_x, xhi, xlo);
  nce_main<<<GRID_MAIN, 128, 0, stream>>>(features, xhi, xlo, out + 1, partial);
  nce_finalize<<<1, 256, 0, stream>>>(features, scores, targets, ws_x, partial,
                                      out);
}
